// LocalMultiHeadedAttention_11441792876631
// MI455X (gfx1250) — matrix operand order repaired and checked by reading, not compiled, not run on hardware
//
#include <hip/hip_runtime.h>

// ---------------------------------------------------------------------------
// LocalMultiHeadedAttention for MI455X (gfx1250), wave32 + WMMA bf16.
// B=8, S=4096, D=768, H=12, dh=64, window=5 (offsets -2..+2, zero padded).
// Algebraic shortcut: k/v projections of shifted windows == shifted rows of
// x@Wk+bk (bias-only rows at boundaries), so we run 4 GEMMs + tiny attention.
// ---------------------------------------------------------------------------

#define BATCH 8
#define SEQ   4096
#define DIM   768
#define HEADS 12
#define DHEAD 64
#define KWIN  5
#define MTOT  (BATCH * SEQ)   // 32768 rows

typedef __attribute__((ext_vector_type(16))) __bf16 v16bf;
typedef __attribute__((ext_vector_type(8)))  __bf16 v8bf;
typedef __attribute__((ext_vector_type(4)))  __bf16 v4bf;
typedef __attribute__((ext_vector_type(8)))  float  v8f;

// ---------------------------------------------------------------------------
// fp32 -> bf16 conversion (vectorized, 4 elems/thread)
// ---------------------------------------------------------------------------
__global__ __launch_bounds__(256) void f32_to_bf16_kernel(
    const float* __restrict__ in, __bf16* __restrict__ out, int n4) {
  int i = blockIdx.x * blockDim.x + threadIdx.x;
  if (i >= n4) return;
  float4 v = ((const float4*)in)[i];
  v4bf o;
  o[0] = (__bf16)v.x; o[1] = (__bf16)v.y; o[2] = (__bf16)v.z; o[3] = (__bf16)v.w;
  ((v4bf*)out)[i] = o;
}

// ---------------------------------------------------------------------------
// W[k,n] (fp32, DIMxDIM) -> Wt[n,k] (bf16)  : coalesced writes
// ---------------------------------------------------------------------------
__global__ __launch_bounds__(256) void transpose_to_bf16_kernel(
    const float* __restrict__ W, __bf16* __restrict__ Wt) {
  int idx = blockIdx.x * blockDim.x + threadIdx.x;   // n*DIM + k
  if (idx >= DIM * DIM) return;
  int n = idx / DIM;
  int k = idx - n * DIM;
  Wt[idx] = (__bf16)W[k * DIM + n];
}

// ---------------------------------------------------------------------------
// WMMA GEMM:  C[M,N] = A[M,K] * Bt[N,K]^T + bias[N] (+ residual[M,N])
// block = 256 threads = 8 waves, arranged 2 (row) x 4 (col);
// each wave computes a 32x64 C tile = 2x4 fragments of 16x16,
// K-loop step 32 with v_wmma_f32_16x16x32_bf16 (8 WMMA per 12 b128 loads).
// Fragment layouts per CDNA5 ISA 7.12.2 (16-bit A 16x32; B mirrors with
// lane-half h = k 8h..8h+7, 16+8h..23+8h, 2 consecutive k per VGPR).
// ---------------------------------------------------------------------------
__device__ __forceinline__ v16bf cat8(v8bf lo, v8bf hi) {
  v16bf r;
#pragma unroll
  for (int i = 0; i < 8; ++i) { r[i] = lo[i]; r[i + 8] = hi[i]; }
  return r;
}

__global__ __launch_bounds__(256) void wmma_gemm_kernel(
    const __bf16* __restrict__ A,    // M x K (row major)
    const __bf16* __restrict__ Bt,   // N x K (row major) == B transposed
    const float*  __restrict__ bias, // N
    const float*  __restrict__ residual, // M x N or nullptr
    float* __restrict__ C,           // M x N
    int M, int N, int K) {
  const int lane = threadIdx.x & 31;
  const int wave = threadIdx.x >> 5;
  const int row0 = blockIdx.y * 64 + (wave >> 2) * 32;   // 2 wave-rows
  const int col0 = blockIdx.x * 256 + (wave & 3) * 64;   // 4 wave-cols
  const int r  = lane & 15;
  const int hi = lane >> 4;

  v8f acc[2][4] = {};

  // A fragment: row = r (+0/+16), chunks of 8 bf16 at k = koff and koff+16
  const __bf16* a0p = A + (size_t)(row0 + r) * K + hi * 8;
  const __bf16* a1p = A + (size_t)(row0 + 16 + r) * K + hi * 8;
  // B fragments: col = r (+0/16/32/48), chunks of 8 bf16 at k = hi*8 and hi*8+16
  const __bf16* bp[4];
#pragma unroll
  for (int cf = 0; cf < 4; ++cf)
    bp[cf] = Bt + (size_t)(col0 + cf * 16 + r) * K + hi * 8;

#pragma unroll 2
  for (int kk = 0; kk < K; kk += 32) {
    v16bf a0 = cat8(*(const v8bf*)(a0p + kk), *(const v8bf*)(a0p + kk + 16));
    v16bf a1 = cat8(*(const v8bf*)(a1p + kk), *(const v8bf*)(a1p + kk + 16));
    v16bf b[4];
#pragma unroll
    for (int cf = 0; cf < 4; ++cf)
      b[cf] = cat8(*(const v8bf*)(bp[cf] + kk), *(const v8bf*)(bp[cf] + kk + 16));

#pragma unroll
    for (int cf = 0; cf < 4; ++cf) {
      acc[0][cf] = __builtin_amdgcn_wmma_f32_16x16x32_bf16(
          false, a0, false, b[cf], (short)0, acc[0][cf], false, false);
      acc[1][cf] = __builtin_amdgcn_wmma_f32_16x16x32_bf16(
          false, a1, false, b[cf], (short)0, acc[1][cf], false, false);
    }
  }

  // C layout: col = col0 + cf*16 + (lane&15); row = row0 + rf*16 + hi*8 + j
#pragma unroll
  for (int rf = 0; rf < 2; ++rf) {
#pragma unroll
    for (int cf = 0; cf < 4; ++cf) {
      const int col = col0 + cf * 16 + r;
      const float bv = bias[col];
#pragma unroll
      for (int j = 0; j < 8; ++j) {
        const int row = row0 + rf * 16 + hi * 8 + j;
        const size_t idx = (size_t)row * N + col;
        float v = acc[rf][cf][j] + bv;
        if (residual) v += residual[idx];
        C[idx] = v;
      }
    }
  }
}

// ---------------------------------------------------------------------------
// Local windowed attention: one thread per (token m, head h).
// scores_i = dot(q, k_row(s+i-2)) / 8 ; OOB rows == bias-only (bk / bv).
// Output att written in bf16 (feeds O projection GEMM).
// ---------------------------------------------------------------------------
__global__ __launch_bounds__(256) void local_attn_kernel(
    const float* __restrict__ Q, const float* __restrict__ Km,
    const float* __restrict__ V,
    const float* __restrict__ bk, const float* __restrict__ bv,
    __bf16* __restrict__ att) {
  int gid = blockIdx.x * blockDim.x + threadIdx.x;
  if (gid >= MTOT * HEADS) return;
  const int m = gid / HEADS;
  const int h = gid - m * HEADS;
  const int s = m & (SEQ - 1);

  const float4* qp = (const float4*)(Q + (size_t)m * DIM + h * DHEAD);
  float4 qv[DHEAD / 4];
#pragma unroll
  for (int e = 0; e < DHEAD / 4; ++e) qv[e] = qp[e];

  float sc[KWIN];
#pragma unroll
  for (int i = 0; i < KWIN; ++i) {
    const int src = s + i - 2;
    const bool oob = (src < 0) | (src >= SEQ);
    const float4* kp = oob
        ? (const float4*)(bk + h * DHEAD)
        : (const float4*)(Km + (size_t)(m + i - 2) * DIM + h * DHEAD);
    float d = 0.f;
#pragma unroll
    for (int e = 0; e < DHEAD / 4; ++e) {
      float4 kv = kp[e];
      d += qv[e].x * kv.x + qv[e].y * kv.y + qv[e].z * kv.z + qv[e].w * kv.w;
    }
    sc[i] = d * 0.125f;  // / sqrt(64)
  }

  float mx = sc[0];
#pragma unroll
  for (int i = 1; i < KWIN; ++i) mx = fmaxf(mx, sc[i]);
  float sum = 0.f;
#pragma unroll
  for (int i = 0; i < KWIN; ++i) { sc[i] = __expf(sc[i] - mx); sum += sc[i]; }
  const float inv = 1.f / sum;

  float4 acc[DHEAD / 4] = {};
#pragma unroll
  for (int i = 0; i < KWIN; ++i) {
    const int src = s + i - 2;
    const bool oob = (src < 0) | (src >= SEQ);
    const float4* vp = oob
        ? (const float4*)(bv + h * DHEAD)
        : (const float4*)(V + (size_t)(m + i - 2) * DIM + h * DHEAD);
    const float p = sc[i] * inv;
#pragma unroll
    for (int e = 0; e < DHEAD / 4; ++e) {
      float4 vv = vp[e];
      acc[e].x += p * vv.x; acc[e].y += p * vv.y;
      acc[e].z += p * vv.z; acc[e].w += p * vv.w;
    }
  }

  v4bf* op = (v4bf*)(att + (size_t)m * DIM + h * DHEAD);
#pragma unroll
  for (int e = 0; e < DHEAD / 4; ++e) {
    v4bf o;
    o[0] = (__bf16)acc[e].x; o[1] = (__bf16)acc[e].y;
    o[2] = (__bf16)acc[e].z; o[3] = (__bf16)acc[e].w;
    op[e] = o;
  }
}

// ---------------------------------------------------------------------------
// In-place LayerNorm over last dim (768), one 256-thread block per row.
// var is population variance (matches jnp.var), eps inside sqrt.
// ---------------------------------------------------------------------------
__global__ __launch_bounds__(256) void layernorm_kernel(
    float* __restrict__ y, const float* __restrict__ gamma,
    const float* __restrict__ beta) {
  const int m = blockIdx.x;
  const int t = threadIdx.x;
  float* row = y + (size_t)m * DIM;
  float v0 = row[t], v1 = row[t + 256], v2 = row[t + 512];

  __shared__ float s1[256];
  __shared__ float s2[256];
  s1[t] = v0 + v1 + v2;
  s2[t] = v0 * v0 + v1 * v1 + v2 * v2;
  __syncthreads();
  for (int off = 128; off > 0; off >>= 1) {
    if (t < off) { s1[t] += s1[t + off]; s2[t] += s2[t + off]; }
    __syncthreads();
  }
  const float mean = s1[0] * (1.f / DIM);
  const float var  = s2[0] * (1.f / DIM) - mean * mean;
  const float inv  = rsqrtf(var + 1e-5f);

  row[t]       = (v0 - mean) * inv * gamma[t]       + beta[t];
  row[t + 256] = (v1 - mean) * inv * gamma[t + 256] + beta[t + 256];
  row[t + 512] = (v2 - mean) * inv * gamma[t + 512] + beta[t + 512];
}

// ---------------------------------------------------------------------------
// Host-side orchestration
// ---------------------------------------------------------------------------
extern "C" void kernel_launch(void* const* d_in, const int* in_sizes, int n_in,
                              void* d_out, int out_size, void* d_ws, size_t ws_size,
                              hipStream_t stream) {
  const float* x     = (const float*)d_in[0];
  const float* Wq    = (const float*)d_in[1];
  const float* bq    = (const float*)d_in[2];
  const float* Wk    = (const float*)d_in[3];
  const float* bk    = (const float*)d_in[4];
  const float* Wv    = (const float*)d_in[5];
  const float* bv    = (const float*)d_in[6];
  const float* Wo    = (const float*)d_in[7];
  const float* bo    = (const float*)d_in[8];
  const float* gamma = (const float*)d_in[9];
  const float* beta  = (const float*)d_in[10];

  // workspace layout (all offsets 256B aligned)
  char* ws = (char*)d_ws;
  const size_t szXbf   = (size_t)MTOT * DIM * sizeof(__bf16);   // 48 MB
  const size_t szWt    = (size_t)DIM * DIM * sizeof(__bf16);    // 1.125 MB
  const size_t szF32   = (size_t)MTOT * DIM * sizeof(float);    // 96 MB

  __bf16* x_bf  = (__bf16*)(ws);
  __bf16* Wq_t  = (__bf16*)(ws + szXbf);
  __bf16* Wk_t  = (__bf16*)(ws + szXbf + szWt);
  __bf16* Wv_t  = (__bf16*)(ws + szXbf + 2 * szWt);
  __bf16* Wo_t  = (__bf16*)(ws + szXbf + 3 * szWt);
  float*  Qf    = (float*)(ws + szXbf + 4 * szWt);
  float*  Kf    = (float*)(ws + szXbf + 4 * szWt + szF32);
  float*  Vf    = (float*)(ws + szXbf + 4 * szWt + 2 * szF32);
  __bf16* attbf = (__bf16*)(ws + szXbf + 4 * szWt + 3 * szF32);
  float*  y     = (float*)d_out;   // pre-LN result lives in d_out, LN in place

  // 1) convert x to bf16
  {
    const int n4 = MTOT * DIM / 4;
    f32_to_bf16_kernel<<<(n4 + 255) / 256, 256, 0, stream>>>(x, x_bf, n4);
  }
  // 2) transpose + convert weights
  {
    const int nb = (DIM * DIM + 255) / 256;
    transpose_to_bf16_kernel<<<nb, 256, 0, stream>>>(Wq, Wq_t);
    transpose_to_bf16_kernel<<<nb, 256, 0, stream>>>(Wk, Wk_t);
    transpose_to_bf16_kernel<<<nb, 256, 0, stream>>>(Wv, Wv_t);
    transpose_to_bf16_kernel<<<nb, 256, 0, stream>>>(Wo, Wo_t);
  }
  // 3) Q/K/V projections: (32768x768) @ (768x768) + bias
  {
    dim3 grid(DIM / 256, MTOT / 64);   // (3, 512)
    dim3 block(256);
    wmma_gemm_kernel<<<grid, block, 0, stream>>>(x_bf, Wq_t, bq, nullptr, Qf,
                                                 MTOT, DIM, DIM);
    wmma_gemm_kernel<<<grid, block, 0, stream>>>(x_bf, Wk_t, bk, nullptr, Kf,
                                                 MTOT, DIM, DIM);
    wmma_gemm_kernel<<<grid, block, 0, stream>>>(x_bf, Wv_t, bv, nullptr, Vf,
                                                 MTOT, DIM, DIM);
  }
  // 4) local windowed attention -> att (bf16)
  {
    const int nthr = MTOT * HEADS;
    local_attn_kernel<<<nthr / 256, 256, 0, stream>>>(Qf, Kf, Vf, bk, bv, attbf);
  }
  // 5) output projection + residual: y = att @ Wo + bo + x   (into d_out)
  {
    dim3 grid(DIM / 256, MTOT / 64);
    wmma_gemm_kernel<<<grid, dim3(256), 0, stream>>>(attbf, Wo_t, bo, x, y,
                                                     MTOT, DIM, DIM);
  }
  // 6) LayerNorm in place on d_out
  layernorm_kernel<<<MTOT, 256, 0, stream>>>(y, gamma, beta);
}
